// SimpleGCN_90666759618832
// MI455X (gfx1250) — matrix operand order repaired and checked by reading, not compiled, not run on hardware
//
#include <hip/hip_runtime.h>
#include <hip/hip_bf16.h>

typedef __bf16 bf16;
typedef __attribute__((ext_vector_type(16))) __bf16 v16bf;
typedef __attribute__((ext_vector_type(8)))  __bf16 v8bf;
typedef __attribute__((ext_vector_type(8)))  float  v8f;

// ---------------- utility kernels ----------------

__global__ void zero_f32_kernel(float* __restrict__ p, long n) {
  long i = (long)blockIdx.x * blockDim.x + threadIdx.x;
  long stride = (long)gridDim.x * blockDim.x;
  for (; i < n; i += stride) p[i] = 0.0f;
}

__global__ void deg_count_kernel(const int* __restrict__ dst, float* __restrict__ deg, int E) {
  int i = blockIdx.x * blockDim.x + threadIdx.x;
  if (i < E) atomicAdd(&deg[dst[i]], 1.0f);
}

// deg -> deg^-1/2 in place; +1 accounts for the self-loop, so deg >= 1 always.
__global__ void dis_kernel(float* __restrict__ deg, int n) {
  int i = blockIdx.x * blockDim.x + threadIdx.x;
  if (i < n) deg[i] = rsqrtf(deg[i] + 1.0f);
}

__global__ void cvt_bf16_kernel(const float* __restrict__ in, bf16* __restrict__ out, long n) {
  long i = (long)blockIdx.x * blockDim.x + threadIdx.x;
  long stride = (long)gridDim.x * blockDim.x;
  for (; i < n; i += stride) out[i] = (bf16)in[i];
}

// W: [K][N] row-major fp32  ->  Wt: [N][K] bf16 (B^T layout for WMMA B operand)
__global__ void transpose_bf16_kernel(const float* __restrict__ W, bf16* __restrict__ Wt,
                                      int K, int N) {
  int idx = blockIdx.x * blockDim.x + threadIdx.x;
  if (idx < K * N) {
    int k = idx / N, nn = idx % N;
    Wt[(long)nn * K + k] = (bf16)W[idx];
  }
}

// ---------------- WMMA GEMM: C[M,N] = A[M,K] * Bt[N,K]^T ----------------
// One wave per 16x32 output tile. grid = (M/16, N/32), block = 32 (EXEC all ones).
__global__ __launch_bounds__(32)
void gemm_bf16_wmma_kernel(const bf16* __restrict__ A, const bf16* __restrict__ Bt,
                           float* __restrict__ C, int N, int K) {
  const int lane = threadIdx.x;        // 0..31
  const int half = lane >> 4;          // 0 or 1
  const int r    = lane & 15;
  const long m0 = (long)blockIdx.x * 16;
  const int  n0 = blockIdx.y * 32;

  const bf16* Arow = A + (m0 + r) * (long)K;
  const bf16* B0   = Bt + (long)(n0 + r) * K;
  const bf16* B1   = Bt + (long)(n0 + 16 + r) * K;

  v8f c0 = {}, c1 = {};

  for (int k0 = 0; k0 < K; k0 += 32) {
    union { v16bf v; v8bf h[2]; } a, b0, b1;
    // A k order used here (same as B, not the ISA table order): lane (half,r) holds 16 contiguous K, K offset half*16
    a.h[0] = *(const v8bf*)(Arow + k0 + half * 16);
    a.h[1] = *(const v8bf*)(Arow + k0 + half * 16 + 8);
    // B k order used here (same as A): lane (half,r) holds 16 contiguous K at column n0+r, K offset half*16
    const int kb = k0 + half * 16;
    b0.h[0] = *(const v8bf*)(B0 + kb);
    b0.h[1] = *(const v8bf*)(B0 + kb + 8);
    b1.h[0] = *(const v8bf*)(B1 + kb);
    b1.h[1] = *(const v8bf*)(B1 + kb + 8);

    c0 = __builtin_amdgcn_wmma_f32_16x16x32_bf16(false, a.v, false, b0.v,
                                                 (short)0, c0, false, false);
    c1 = __builtin_amdgcn_wmma_f32_16x16x32_bf16(false, a.v, false, b1.v,
                                                 (short)0, c1, false, false);
  }

  // C/D layout: VGPR v -> row m0 + v + 8*half, col n0 + r (+16 for second tile)
#pragma unroll
  for (int v = 0; v < 8; ++v) {
    long row = m0 + v + half * 8;
    C[row * N + n0 + r]      = c0[v];
    C[row * N + n0 + 16 + r] = c1[v];
  }
}

// ---------------- edge scatter: acc[dst] += h[src] * dis[src]*dis[dst] ----------------
// one block per edge, one thread per feature
__global__ void scatter_kernel(const float* __restrict__ h, const int* __restrict__ src,
                               const int* __restrict__ dst, const float* __restrict__ dis,
                               float* __restrict__ acc, int N) {
  const int e = blockIdx.x;
  const int f = threadIdx.x;
  const int s = src[e];
  const int d = dst[e];
  const float w = dis[s] * dis[d];
  atomicAdd(&acc[(long)d * N + f], h[(long)s * N + f] * w);
}

// layer-1 finalize: add self-loop term + bias, ReLU, emit bf16 for next GEMM
__global__ void finalize1_kernel(const float* __restrict__ acc, const float* __restrict__ h,
                                 const float* __restrict__ dis, const float* __restrict__ bias,
                                 bf16* __restrict__ outb, long total, int N) {
  long i = (long)blockIdx.x * blockDim.x + threadIdx.x;
  long stride = (long)gridDim.x * blockDim.x;
  for (; i < total; i += stride) {
    long row = i / N;
    int f = (int)(i - row * N);
    float di = dis[row];
    float v = acc[i] + h[i] * di * di + bias[f];
    outb[i] = (bf16)fmaxf(v, 0.0f);
  }
}

// layer-2 finalize (in place on d_out which holds the scatter accumulation)
__global__ void finalize2_kernel(float* __restrict__ acc, const float* __restrict__ h,
                                 const float* __restrict__ dis, const float* __restrict__ bias,
                                 long total, int N) {
  long i = (long)blockIdx.x * blockDim.x + threadIdx.x;
  long stride = (long)gridDim.x * blockDim.x;
  for (; i < total; i += stride) {
    long row = i / N;
    int f = (int)(i - row * N);
    float di = dis[row];
    acc[i] = acc[i] + h[i] * di * di + bias[f];
  }
}

// ---------------- launcher ----------------

extern "C" void kernel_launch(void* const* d_in, const int* in_sizes, int n_in,
                              void* d_out, int out_size, void* d_ws, size_t ws_size,
                              hipStream_t stream) {
  const float* x  = (const float*)d_in[0];
  const int*   ei = (const int*)d_in[1];
  const float* W1 = (const float*)d_in[2];
  const float* b1 = (const float*)d_in[3];
  const float* W2 = (const float*)d_in[4];
  const float* b2 = (const float*)d_in[5];
  float* out = (float*)d_out;

  const int F0 = 256, F1 = 256, F2 = 128;
  const int n = in_sizes[0] / F0;      // 50000 nodes
  const int E = in_sizes[1] / 2;       // 800000 edges
  const int* src = ei;
  const int* dst = ei + E;

  // workspace bump allocator (256B aligned)
  char* ws = (char*)d_ws;
  size_t off = 0;
  auto alloc = [&](size_t bytes) -> void* {
    void* p = ws + off;
    off = (off + bytes + 255) & ~(size_t)255;
    return p;
  };
  float* dis  = (float*)alloc((size_t)n * 4);             // deg -> deg^-1/2
  bf16*  xb   = (bf16*)alloc((size_t)n * F0 * 2);         // bf16 x; reused as h1b later
  bf16*  w1t  = (bf16*)alloc((size_t)F1 * F0 * 2);        // W1^T bf16 [N][K]
  bf16*  w2t  = (bf16*)alloc((size_t)F2 * F1 * 2);        // W2^T bf16 [N][K]
  float* h1   = (float*)alloc((size_t)n * F1 * 4);        // x @ W1
  float* acc1 = (float*)alloc((size_t)n * F1 * 4);        // layer-1 scatter; reused as h2
  bf16*  h1b  = xb;                                       // alias: xb dead after GEMM1
  float* h2   = acc1;                                     // alias: acc1 dead after finalize1

  const long nF1 = (long)n * F1;
  const long nF2 = (long)n * F2;

  // --- normalization coefficients ---
  zero_f32_kernel<<<(n + 255) / 256, 256, 0, stream>>>(dis, n);
  deg_count_kernel<<<(E + 255) / 256, 256, 0, stream>>>(dst, dis, E);
  dis_kernel<<<(n + 255) / 256, 256, 0, stream>>>(dis, n);

  // --- bf16 staging ---
  cvt_bf16_kernel<<<4096, 256, 0, stream>>>(x, xb, (long)n * F0);
  transpose_bf16_kernel<<<(F0 * F1 + 255) / 256, 256, 0, stream>>>(W1, w1t, F0, F1);
  transpose_bf16_kernel<<<(F1 * F2 + 255) / 256, 256, 0, stream>>>(W2, w2t, F1, F2);

  // --- layer 1: GEMM, scatter, finalize(+ReLU, ->bf16) ---
  gemm_bf16_wmma_kernel<<<dim3(n / 16, F1 / 32), 32, 0, stream>>>(xb, w1t, h1, F1, F0);
  zero_f32_kernel<<<8192, 256, 0, stream>>>(acc1, nF1);
  scatter_kernel<<<E, F1, 0, stream>>>(h1, src, dst, dis, acc1, F1);
  finalize1_kernel<<<8192, 256, 0, stream>>>(acc1, h1, dis, b1, h1b, nF1, F1);

  // --- layer 2: GEMM, scatter into d_out, finalize in place ---
  gemm_bf16_wmma_kernel<<<dim3(n / 16, F2 / 32), 32, 0, stream>>>(h1b, w2t, h2, F2, F1);
  zero_f32_kernel<<<8192, 256, 0, stream>>>(out, nF2);
  scatter_kernel<<<E, F2, 0, stream>>>(h2, src, dst, dis, out, F2);
  finalize2_kernel<<<8192, 256, 0, stream>>>(out, h2, dis, b2, nF2, F2);
}
